// LinformerAttention_39865886441712
// MI455X (gfx1250) — matrix operand order repaired and checked by reading, not compiled, not run on hardware
//
#include <hip/hip_runtime.h>

// Problem constants (from the reference)
#define B_  4
#define L_  4096
#define D_  1024
#define H_  16
#define HD_ 64
#define P_  128

typedef __attribute__((ext_vector_type(16))) __bf16       bf16x16;
typedef __attribute__((ext_vector_type(2)))  __bf16       bf16x2;
typedef __attribute__((ext_vector_type(8)))  float        floatx8;
typedef __attribute__((ext_vector_type(4)))  unsigned int uint4v;
typedef __attribute__((ext_vector_type(2)))  unsigned int uint2v;
typedef __attribute__((ext_vector_type(4)))  unsigned int u32x4;
typedef __attribute__((ext_vector_type(8)))  int          i32x8;
typedef __attribute__((ext_vector_type(4)))  int          i32x4;

union Frag { bf16x16 v; uint4v q[2]; };

__device__ inline unsigned short f2bf(float f) {
  unsigned int u = __float_as_uint(f);
  u += 0x7FFFu + ((u >> 16) & 1u);           // round-to-nearest-even
  return (unsigned short)(u >> 16);
}
__device__ inline float bf2f(unsigned short h) {
  return __uint_as_float(((unsigned int)h) << 16);
}

// Packed f32x2 -> bf16x2 conversion: single v_cvt_pk_bf16_f32 when available.
#if __has_builtin(__builtin_amdgcn_cvt_pk_bf16_f32)
__device__ inline unsigned pack_bf16(float lo, float hi) {
  union { bf16x2 v; unsigned u; } t;
  t.v = __builtin_amdgcn_cvt_pk_bf16_f32(lo, hi);
  return t.u;
}
#else
__device__ inline unsigned pack_bf16(float lo, float hi) {
  return (unsigned)f2bf(lo) | ((unsigned)f2bf(hi) << 16);
}
#endif

__device__ inline floatx8 fzero8() {
  floatx8 z = {0.f,0.f,0.f,0.f,0.f,0.f,0.f,0.f};
  return z;
}
__device__ inline floatx8 wmma_bf16(const Frag& a, const Frag& b, floatx8 c) {
  // D = A(16x32 bf16) x B(32x16 bf16) + C(16x16 f32)
  return __builtin_amdgcn_wmma_f32_16x16x32_bf16(
      false, a.v, false, b.v, (short)0, c, false, false);
}

// ---------------------------------------------------------------------------
// Tensor Data Mover: async 2D tile load (128 rows x 64 bf16, contiguous in
// memory) into LDS with 8-bf16 row padding (LDS row stride = 72 bf16).
//   pad_interval=4 -> pad after every 32 stored DWORDs (one 128B row)
//   pad_amount=3   -> insert 4 DWORDs (16B) of padding
// D# group0: count=1 | lds_addr | global_addr | type=2
// D# group1: data_size=2B, tensor_dim0=64, tensor_dim1=128, tile 64x128,
//            dim0_stride=64
// This toolchain uses the 6-arg builtin form:
//   (u32x4 g0, i32x8 g1, i32x4 g2, i32x4 g3, i32x8 pad, i32 cpol)
// ---------------------------------------------------------------------------
#if __has_builtin(__builtin_amdgcn_tensor_load_to_lds) && \
    __has_builtin(__builtin_amdgcn_s_wait_tensorcnt)
#define HAVE_TDM 1
__device__ inline void tdm_load_tile_128x64(unsigned lds_off,
                                            const unsigned short* gptr) {
  unsigned long long ga = (unsigned long long)(const void*)gptr;
  u32x4 g0;
  g0.x = 1u;                                            // count=1 (valid D#)
  g0.y = lds_off;                                       // LDS byte address
  g0.z = (unsigned)(ga & 0xFFFFFFFFull);                // global_addr[31:0]
  g0.w = (unsigned)((ga >> 32) & 0x01FFFFFFull)         // global_addr[56:32]
       | (2u << 30);                                    // type=2 ("image")
  i32x8 g1;
  g1[0] = (int)((1u << 16)      // data_size = 2 bytes
              | (1u << 20)      // pad_enable
              | (4u << 22)      // pad_interval: 32 DWORDs
              | (3u << 25));    // pad_amount: 4 DWORDs
  g1[1] = (int)(64u  << 16);    // tensor_dim0 = 64 elements
  g1[2] = (int)(128u << 16);    // tensor_dim1 = 128
  g1[3] = (int)(64u  << 16);    // tile_dim0 = 64
  g1[4] = 128;                  // tile_dim1 = 128
  g1[5] = 64;                   // tensor_dim0_stride = 64 elements
  g1[6] = 0;
  g1[7] = 0;
  i32x4 z4 = {0, 0, 0, 0};      // groups 2/3 unused (2D tensor)
  i32x8 z8 = {0, 0, 0, 0, 0, 0, 0, 0};
  __builtin_amdgcn_tensor_load_to_lds(g0, g1, z4, z4, z8, 0);
}
#endif

// ---------------------------------------------------------------------------
// Kernel 1: fused QKV projection (x @ W{q,k,v}.T) + RoPE on Q,K.
// Grid: (8 col-tiles * 3 matrices, 128 row-tiles), 256 threads (8 waves).
// Block tile: 128(M) x 128(N), K-step 32 (bf16 WMMA K).
// Output layout: (B, H, L, HD) bf16.
// ---------------------------------------------------------------------------
__global__ __launch_bounds__(256) void qkv_rope_kernel(
    const float* __restrict__ x,
    const float* __restrict__ Wq, const float* __restrict__ Wk,
    const float* __restrict__ Wv,
    const float* __restrict__ fcos, const float* __restrict__ fsin,
    unsigned short* __restrict__ Qb, unsigned short* __restrict__ Kb,
    unsigned short* __restrict__ Vb)
{
  const int mat  = blockIdx.x >> 3;             // 0=Q, 1=K, 2=V
  const int col0 = (blockIdx.x & 7) * 128;
  const int row0 = blockIdx.y * 128;
  const float* W = (mat == 0) ? Wq : (mat == 1) ? Wk : Wv;
  unsigned short* Ob = (mat == 0) ? Qb : (mat == 1) ? Kb : Vb;

  // row stride 40 bf16 (80B, multiple of 16B, conflict-skewed)
  __shared__ __align__(16) unsigned short As[128 * 40];
  __shared__ __align__(16) unsigned short Bs[128 * 40];

  const int tid  = threadIdx.x;
  const int lane = tid & 31;
  const int w    = tid >> 5;
  const int wm   = w >> 1;                      // 0..3 : 32 rows each
  const int wn   = w & 1;                       // 0..1 : 64 cols each

  floatx8 acc[2][4];
  #pragma unroll
  for (int i = 0; i < 2; ++i)
    #pragma unroll
    for (int j = 0; j < 4; ++j) acc[i][j] = fzero8();

  const int fr = lane & 15;
  const int ak = (lane < 16) ? 0 : 8;           // A-frag K base (bf16 16x32 layout)
  const int bk = (lane < 16) ? 0 : 8;           // B-frag K base (bf16 32x16 layout)

  for (int kt = 0; kt < D_ / 32; ++kt) {
    const int kb = kt * 32;
    if (kt + 1 < D_ / 32) {                     // CDNA5 global_prefetch of next K-tile
      __builtin_prefetch(x + (size_t)(row0 + (tid >> 1)) * D_ + kb + 32, 0, 1);
      __builtin_prefetch(W + (size_t)(col0 + (tid >> 1)) * D_ + kb + 32, 0, 1);
    }
    // Stage A (x rows) and B (W rows) tiles as bf16 into LDS.
    #pragma unroll
    for (int i = 0; i < 4; ++i) {
      int f = i * 256 + tid;                    // float4 index 0..1023
      int r = f >> 3, c4 = f & 7;
      float4 a4 = *(const float4*)(x + (size_t)(row0 + r) * D_ + kb + c4 * 4);
      uint2v pa;
      pa.x = pack_bf16(a4.x, a4.y);
      pa.y = pack_bf16(a4.z, a4.w);
      *(uint2v*)&As[r * 40 + c4 * 4] = pa;
      float4 b4 = *(const float4*)(W + (size_t)(col0 + r) * D_ + kb + c4 * 4);
      uint2v pb;
      pb.x = pack_bf16(b4.x, b4.y);
      pb.y = pack_bf16(b4.z, b4.w);
      *(uint2v*)&Bs[r * 40 + c4 * 4] = pb;
    }
    __syncthreads();

    Frag a[2], bfm[4];
    #pragma unroll
    for (int i = 0; i < 2; ++i) {               // A-frag: lane holds row, K {ak..ak+7, ak+16..}
      const unsigned short* p = &As[(wm * 32 + i * 16 + fr) * 40 + ak];
      a[i].q[0] = *(const uint4v*)p;
      a[i].q[1] = *(const uint4v*)(p + 16);
    }
    #pragma unroll
    for (int j = 0; j < 4; ++j) {               // B-frag: lane holds col n, K {bk..bk+7, bk+16..}
      const unsigned short* p = &Bs[(wn * 64 + j * 16 + fr) * 40 + bk];
      bfm[j].q[0] = *(const uint4v*)p;
      bfm[j].q[1] = *(const uint4v*)(p + 16);
    }
    #pragma unroll
    for (int i = 0; i < 2; ++i)
      #pragma unroll
      for (int j = 0; j < 4; ++j)
        acc[i][j] = wmma_bf16(a[i], bfm[j], acc[i][j]);
    __syncthreads();
  }

  // Epilogue: RoPE (Q,K only) fused via cross-lane exchange, store bf16 (B,H,L,HD).
  const int bidx    = row0 >> 12;               // batch (128 rows never cross batch)
  const int halfadd = (lane < 16) ? 0 : 8;      // C layout: VGPR r -> row r / r+8
  #pragma unroll
  for (int i = 0; i < 2; ++i)
    #pragma unroll
    for (int j = 0; j < 4; ++j) {
      int gn = col0 + wn * 64 + j * 16 + fr;
      int h = gn >> 6, hd = gn & 63;
      #pragma unroll
      for (int r = 0; r < 8; ++r) {
        int l = (row0 + wm * 32 + i * 16 + r + halfadd) & (L_ - 1);
        float v = acc[i][j][r];
        if (mat < 2) {
          float c  = fcos[l * 32 + (hd >> 1)];
          float s  = fsin[l * 32 + (hd >> 1)];
          float pv = __shfl_xor(v, 1, 32);      // even/odd rotation partner (adjacent col)
          v = ((hd & 1) == 0) ? (v * c - pv * s) : (pv * s + v * c);
        }
        Ob[((size_t)(bidx * H_ + h) * L_ + l) * HD_ + hd] = f2bf(v);
      }
    }
}

// ---------------------------------------------------------------------------
// Kernel 2: K_proj / V_proj.  tril() kills every column l > p, so only the
// first 128 sequence positions contribute: per (b,h) it is a 128x128x64
// masked GEMM (~0.27 GFLOP total) -> plain FMA is fine.
// ---------------------------------------------------------------------------
__global__ __launch_bounds__(256) void proj_kernel(
    const float* __restrict__ kpm, const float* __restrict__ vpm,
    const unsigned short* __restrict__ Kb, const unsigned short* __restrict__ Vb,
    unsigned short* __restrict__ Kpb, unsigned short* __restrict__ Vpb)
{
  const int bh  = blockIdx.x;                   // 0..63
  const int tid = threadIdx.x;
  const int p   = tid >> 1;                     // 0..127
  const int hd0 = (tid & 1) * 32;
  const unsigned short* Ksrc = Kb + (size_t)bh * L_ * HD_;
  const unsigned short* Vsrc = Vb + (size_t)bh * L_ * HD_;

  float acck[32], accv[32];
  #pragma unroll
  for (int j = 0; j < 32; ++j) { acck[j] = 0.f; accv[j] = 0.f; }

  for (int l = 0; l <= p; ++l) {                // tril: l <= p only
    float ak = kpm[p * 4096 + l];
    float av = vpm[p * 4096 + l];
    #pragma unroll
    for (int j = 0; j < 32; ++j) {
      float kk = bf2f(Ksrc[l * HD_ + hd0 + j]);
      float vv = bf2f(Vsrc[l * HD_ + hd0 + j]);
      acck[j] += ak * kk;
      accv[j] += av * vv;
    }
  }
  unsigned short* kout = Kpb + (size_t)bh * P_ * HD_ + p * HD_ + hd0;
  unsigned short* vout = Vpb + (size_t)bh * P_ * HD_ + p * HD_ + hd0;
  #pragma unroll
  for (int j = 0; j < 32; ++j) { kout[j] = f2bf(acck[j]); vout[j] = f2bf(accv[j]); }
}

// ---------------------------------------------------------------------------
// Kernel 3: fused attention per (b,h) 128-row block:
//   scores = Q*Kp^T/8  -> mask(p>l) -> softmax (in-register, half-wave shfl
//   reductions) -> weights (bf16, LDS) x Vp -> bf16 output (B, L, D).
// Q and Kp tiles are staged via the Tensor Data Mover (async, padded rows);
// Vp is transposed manually (TDM cannot transpose).
// Grid: (L/128, B*H), 256 threads (8 waves, 16 rows each).
// ---------------------------------------------------------------------------
__global__ __launch_bounds__(256) void attn_kernel(
    const unsigned short* __restrict__ Qb, const unsigned short* __restrict__ Kpb,
    const unsigned short* __restrict__ Vpb, unsigned short* __restrict__ Ob)
{
  const int bh   = blockIdx.y;
  const int row0 = blockIdx.x * 128;
  const int b    = bh >> 4, h = bh & 15;

  // 27136 ushorts = 54.3 KB.  Ws overlays Qs+Kps (dead after scores GEMM).
  __shared__ __align__(16) unsigned short smem[27136];
  unsigned short* Qs  = smem;                   // 128*72
  unsigned short* Kps = smem + 9216;            // 128*72
  unsigned short* Vpt = smem + 18432;           // 64*136  (Vp transposed)
  unsigned short* Ws  = smem;                   // 128*136 (overlay)

  const int tid = threadIdx.x, lane = tid & 31, w = tid >> 5;
  const int fr = lane & 15;
  const int ak = (lane < 16) ? 0 : 8;
  const int bk = (lane < 16) ? 0 : 8;
  const int halfadd = (lane < 16) ? 0 : 8;

  const unsigned short* Qsrc  = Qb  + ((size_t)bh * L_ + row0) * HD_;
  const unsigned short* Kpsrc = Kpb + (size_t)bh * P_ * HD_;
  const unsigned short* Vpsrc = Vpb + (size_t)bh * P_ * HD_;

#ifdef HAVE_TDM
  // Async TDM tile loads: wave 0 pulls the Q tile, wave 1 pulls Kp.
  if (w == 0) {
    tdm_load_tile_128x64((unsigned)(unsigned long long)(void*)Qs, Qsrc);
    __builtin_amdgcn_s_wait_tensorcnt(0);
  } else if (w == 1) {
    tdm_load_tile_128x64((unsigned)(unsigned long long)(void*)Kps, Kpsrc);
    __builtin_amdgcn_s_wait_tensorcnt(0);
  }
#else
  #pragma unroll
  for (int i = 0; i < 4; ++i) {
    int f = i * 256 + tid;                      // 1024 16B chunks
    int r = f >> 3, c = f & 7;
    *(uint4v*)&Qs[r * 72 + c * 8]  = *(const uint4v*)(Qsrc  + r * 64 + c * 8);
    *(uint4v*)&Kps[r * 72 + c * 8] = *(const uint4v*)(Kpsrc + r * 64 + c * 8);
  }
#endif
  // Stage Vp transposed (B-frag of the AV GEMM needs K=p contiguous per hd).
  #pragma unroll 4
  for (int i = 0; i < 16; ++i) {
    int e = i * 256 + tid;                      // 4096 dword chunks
    int p = e >> 5, hd2 = (e & 31) * 2;
    unsigned int u = *(const unsigned int*)(Vpsrc + p * 64 + hd2);
    Vpt[hd2 * 136 + p]       = (unsigned short)(u & 0xFFFFu);
    Vpt[(hd2 + 1) * 136 + p] = (unsigned short)(u >> 16);
  }
  __syncthreads();

  // Scores: wave owns 16 rows x 128 cols (8 N-tiles), K = HD = 64 (2 steps).
  floatx8 sc[8];
  #pragma unroll
  for (int j = 0; j < 8; ++j) sc[j] = fzero8();
  #pragma unroll
  for (int kt = 0; kt < 2; ++kt) {
    Frag a;
    const unsigned short* pa = &Qs[(w * 16 + fr) * 72 + kt * 32 + ak];
    a.q[0] = *(const uint4v*)pa; a.q[1] = *(const uint4v*)(pa + 16);
    #pragma unroll
    for (int j = 0; j < 8; ++j) {
      Frag bb;
      const unsigned short* pb = &Kps[(j * 16 + fr) * 72 + kt * 32 + bk];
      bb.q[0] = *(const uint4v*)pb; bb.q[1] = *(const uint4v*)(pb + 16);
      sc[j] = wmma_bf16(a, bb, sc[j]);
    }
  }

  // Mask + softmax: each row lives in one 16-lane half across 8 accum vectors.
  #pragma unroll
  for (int r = 0; r < 8; ++r) {
    int gl = row0 + w * 16 + r + halfadd;       // global sequence position
    float mx = -3.0e38f;
    #pragma unroll
    for (int j = 0; j < 8; ++j) {
      int p = j * 16 + fr;
      float v = sc[j][r] * 0.125f;              // 1/sqrt(HD)
      if (p > gl) v = -3.0e38f;                 // col_mask: p > l
      sc[j][r] = v;
      mx = fmaxf(mx, v);
    }
    #pragma unroll
    for (int m = 1; m < 16; m <<= 1) mx = fmaxf(mx, __shfl_xor(mx, m, 32));
    float sum = 0.f;
    #pragma unroll
    for (int j = 0; j < 8; ++j) {
      float e = __expf(sc[j][r] - mx);
      sc[j][r] = e; sum += e;
    }
    #pragma unroll
    for (int m = 1; m < 16; m <<= 1) sum += __shfl_xor(sum, m, 32);
    float inv = 1.0f / sum;
    #pragma unroll
    for (int j = 0; j < 8; ++j) sc[j][r] *= inv;
  }
  __syncthreads();                              // Qs/Kps reads done -> safe to overlay

  #pragma unroll
  for (int r = 0; r < 8; ++r) {
    int row = w * 16 + r + halfadd;
    #pragma unroll
    for (int j = 0; j < 8; ++j)
      Ws[row * 136 + j * 16 + fr] = f2bf(sc[j][r]);
  }
  __syncthreads();

  // AV GEMM: 16 rows x 64 cols per wave, K = P = 128 (4 steps).
  floatx8 oc[4];
  #pragma unroll
  for (int j = 0; j < 4; ++j) oc[j] = fzero8();
  #pragma unroll
  for (int kt = 0; kt < 4; ++kt) {
    Frag a;
    const unsigned short* pa = &Ws[(w * 16 + fr) * 136 + kt * 32 + ak];
    a.q[0] = *(const uint4v*)pa; a.q[1] = *(const uint4v*)(pa + 16);
    #pragma unroll
    for (int j = 0; j < 4; ++j) {
      Frag bb;
      const unsigned short* pb = &Vpt[(j * 16 + fr) * 136 + kt * 32 + bk];
      bb.q[0] = *(const uint4v*)pb; bb.q[1] = *(const uint4v*)(pb + 16);
      oc[j] = wmma_bf16(a, bb, oc[j]);
    }
  }
  // Store directly in (B, L, D) layout so the final projection is a plain GEMM.
  #pragma unroll
  for (int j = 0; j < 4; ++j) {
    int hd = j * 16 + fr;
    #pragma unroll
    for (int r = 0; r < 8; ++r) {
      int l = row0 + w * 16 + r + halfadd;
      Ob[((size_t)b * L_ + l) * D_ + h * HD_ + hd] = f2bf(oc[j][r]);
    }
  }
}

// ---------------------------------------------------------------------------
// Kernel 4: output projection attn(bf16, 16384x1024) @ Wo.T -> fp32 d_out.
// Same WMMA skeleton as kernel 1 (A already bf16).
// ---------------------------------------------------------------------------
__global__ __launch_bounds__(256) void oproj_kernel(
    const unsigned short* __restrict__ Ab, const float* __restrict__ Wo,
    float* __restrict__ out)
{
  const int col0 = blockIdx.x * 128;
  const int row0 = blockIdx.y * 128;

  __shared__ __align__(16) unsigned short As[128 * 40];
  __shared__ __align__(16) unsigned short Bs[128 * 40];

  const int tid = threadIdx.x, lane = tid & 31, w = tid >> 5;
  const int wm = w >> 1, wn = w & 1;
  const int fr = lane & 15;
  const int ak = (lane < 16) ? 0 : 8;
  const int bk = (lane < 16) ? 0 : 8;

  floatx8 acc[2][4];
  #pragma unroll
  for (int i = 0; i < 2; ++i)
    #pragma unroll
    for (int j = 0; j < 4; ++j) acc[i][j] = fzero8();

  for (int kt = 0; kt < D_ / 32; ++kt) {
    const int kb = kt * 32;
    if (kt + 1 < D_ / 32) {
      __builtin_prefetch(Ab + (size_t)(row0 + (tid >> 1)) * D_ + kb + 32, 0, 1);
      __builtin_prefetch(Wo + (size_t)(col0 + (tid >> 1)) * D_ + kb + 32, 0, 1);
    }
    #pragma unroll
    for (int i = 0; i < 2; ++i) {               // A tile: already bf16, straight copy
      int f = i * 256 + tid;                    // 512 16B chunks
      int r = f >> 2, c = f & 3;
      *(uint4v*)&As[r * 40 + c * 8] =
          *(const uint4v*)(Ab + (size_t)(row0 + r) * D_ + kb + c * 8);
    }
    #pragma unroll
    for (int i = 0; i < 4; ++i) {               // B tile: Wo fp32 -> bf16
      int f = i * 256 + tid;
      int r = f >> 3, c4 = f & 7;
      float4 b4 = *(const float4*)(Wo + (size_t)(col0 + r) * D_ + kb + c4 * 4);
      uint2v pb;
      pb.x = pack_bf16(b4.x, b4.y);
      pb.y = pack_bf16(b4.z, b4.w);
      *(uint2v*)&Bs[r * 40 + c4 * 4] = pb;
    }
    __syncthreads();

    Frag a[2], bfm[4];
    #pragma unroll
    for (int i = 0; i < 2; ++i) {
      const unsigned short* p = &As[(wm * 32 + i * 16 + fr) * 40 + ak];
      a[i].q[0] = *(const uint4v*)p;
      a[i].q[1] = *(const uint4v*)(p + 16);
    }
    #pragma unroll
    for (int j = 0; j < 4; ++j) {
      const unsigned short* p = &Bs[(wn * 64 + j * 16 + fr) * 40 + bk];
      bfm[j].q[0] = *(const uint4v*)p;
      bfm[j].q[1] = *(const uint4v*)(p + 16);
    }
    #pragma unroll
    for (int i = 0; i < 2; ++i)
      #pragma unroll
      for (int j = 0; j < 4; ++j)
        acc[i][j] = wmma_bf16(a[i], bfm[j], acc[i][j]);
    __syncthreads();
  }

  const int halfadd = (lane < 16) ? 0 : 8;
  #pragma unroll
  for (int i = 0; i < 2; ++i)
    #pragma unroll
    for (int j = 0; j < 4; ++j) {
      int gn = col0 + wn * 64 + j * 16 + fr;
      #pragma unroll
      for (int r = 0; r < 8; ++r) {
        int gm = row0 + wm * 32 + i * 16 + r + halfadd;
        out[(size_t)gm * D_ + gn] = acc[i][j][r];
      }
    }
}

// ---------------------------------------------------------------------------
extern "C" void kernel_launch(void* const* d_in, const int* in_sizes, int n_in,
                              void* d_out, int out_size, void* d_ws, size_t ws_size,
                              hipStream_t stream)
{
  (void)in_sizes; (void)n_in; (void)out_size; (void)ws_size;
  const float* x    = (const float*)d_in[0];
  const float* fcos = (const float*)d_in[1];
  const float* fsin = (const float*)d_in[2];
  const float* Wq   = (const float*)d_in[3];
  const float* Wk   = (const float*)d_in[4];
  const float* Wv   = (const float*)d_in[5];
  const float* Wo   = (const float*)d_in[6];
  const float* kpm  = (const float*)d_in[7];
  const float* vpm  = (const float*)d_in[8];

  char* ws = (char*)d_ws;
  const size_t MB = 1024 * 1024;
  unsigned short* Qb  = (unsigned short*)(ws +  0 * MB);  // 32 MB bf16 (B,H,L,HD)
  unsigned short* Kb  = (unsigned short*)(ws + 32 * MB);  // 32 MB
  unsigned short* Vb  = (unsigned short*)(ws + 64 * MB);  // 32 MB
  unsigned short* Kpb = (unsigned short*)(ws + 96 * MB);  //  1 MB (B,H,P,HD)
  unsigned short* Vpb = (unsigned short*)(ws + 97 * MB);  //  1 MB
  unsigned short* Ab  = (unsigned short*)(ws + 98 * MB);  // 32 MB bf16 (B,L,D)

  qkv_rope_kernel<<<dim3(24, 128), 256, 0, stream>>>(
      x, Wq, Wk, Wv, fcos, fsin, Qb, Kb, Vb);
  proj_kernel<<<dim3(64), 256, 0, stream>>>(kpm, vpm, Kb, Vb, Kpb, Vpb);
  attn_kernel<<<dim3(32, 64), 256, 0, stream>>>(Qb, Kpb, Vpb, Ab);
  oproj_kernel<<<dim3(8, 128), 256, 0, stream>>>(Ab, Wo, (float*)d_out);
}
